// diff_loss_81905026335196
// MI455X (gfx1250) — matrix operand order repaired and checked by reading, not compiled, not run on hardware
//
#include <hip/hip_runtime.h>
#include <hip/hip_bf16.h>
#include <math.h>

typedef __attribute__((ext_vector_type(16))) _Float16 v16h;
typedef __attribute__((ext_vector_type(8)))  _Float16 v8h;
typedef __attribute__((ext_vector_type(8)))  float    v8f;

#define BB   128
#define TT   2048
#define CC   10
#define LL   2048
#define NEG  4
#define HH   128
#define DD   160
#define KK   5
#define LSEL 1024
#define LCMP 512

#define NWIN   768            // 128 anchor + 128 pos + 512 neg
#define TILE   64             // time tile per block
#define H1ROW  136            // h1 row stride (halves) = 68 dwords -> bank stagger
#define XIROW  72             // im2col row stride (halves) = 36 dwords -> bank stagger
#define NKB    20             // conv2: 640/32 k-steps
#define CK1    50             // conv1 K-dim = C*K; col 50 = bias, padded to 64

// ---------------------------------------------------------------- setup ----
__global__ void setup_kernel(const int* __restrict__ samples,
                             const int* __restrict__ start_sel,
                             const int* __restrict__ start_pos,
                             const int* __restrict__ start_neg,
                             int* __restrict__ wdesc, float* __restrict__ Hmax) {
    int i = blockIdx.x * blockDim.x + threadIdx.x;
    if (i < NWIN * HH) Hmax[i] = 0.0f;           // re-zero every call (atomicMax target)
    if (i < NWIN) {
        int kind, row, st, len;
        if (i < BB)            { kind = 0; row = i;        st = start_sel[i];       len = LSEL; }
        else if (i < 2 * BB)   { kind = 0; row = i - BB;   st = start_pos[i - BB];  len = LCMP; }
        else { int j = i - 2 * BB; kind = 1; row = samples[j]; st = start_neg[j];   len = LCMP; }
        wdesc[i * 4 + 0] = kind;
        wdesc[i * 4 + 1] = row;
        wdesc[i * 4 + 2] = st;
        wdesc[i * 4 + 3] = len;
    }
}

// -------------------------------------------- weight -> A-fragment swizzle ----
// A-frag (16x32 f16, wave32): row M = lane%16 ; lane<16 holds K {0..7,16..23},
// lane>=16 holds K {8..15,24..31}.
// conv2: K-dim reordered as kidx = k*128 + c (20 k-blocks).
// conv1: kidx = c*5 + k (<50); kidx==50 carries b1[h] (bias column); rest 0.
__global__ void afrag_kernel(const float* __restrict__ W2, const float* __restrict__ W1,
                             const float* __restrict__ b1,
                             _Float16* __restrict__ Afrag2, _Float16* __restrict__ Afrag1) {
    int g = blockIdx.x * blockDim.x + threadIdx.x;
    if (g < 8 * NKB * 32) {                       // conv2 fragments
        int lane = g & 31;
        int kb   = (g >> 5) % NKB;
        int mt   = g / (NKB * 32);
        int hrow = mt * 16 + (lane & 15);
        #pragma unroll
        for (int e = 0; e < 16; ++e) {
            int kr   = e + ((e < 8) ? 0 : 8) + ((lane < 16) ? 0 : 8);
            int kidx = kb * 32 + kr;
            int k    = kidx >> 7;
            int c    = kidx & 127;
            Afrag2[(size_t)g * 16 + e] = (_Float16)W2[hrow * (HH * KK) + c * KK + k];
        }
    } else if (g < 8 * NKB * 32 + 8 * 2 * 32) {   // conv1 fragments (+bias column)
        int g2   = g - 8 * NKB * 32;              // 0..511
        int lane = g2 & 31;
        int kb   = (g2 >> 5) & 1;
        int mt   = g2 >> 6;
        int hrow = mt * 16 + (lane & 15);
        #pragma unroll
        for (int e = 0; e < 16; ++e) {
            int kr   = e + ((e < 8) ? 0 : 8) + ((lane < 16) ? 0 : 8);
            int kidx = kb * 32 + kr;
            float v  = 0.0f;
            if (kidx < CK1) {
                int c = kidx / KK, k = kidx % KK;
                v = W1[hrow * (CC * KK) + c * KK + k];
            } else if (kidx == CK1) {
                v = b1[hrow];                     // bias folded into GEMM
            }
            Afrag1[(size_t)g2 * 16 + e] = (_Float16)v;
        }
    }
}

// lane-xor max over the 16-lane half (group-of-32 xor swizzles, imm = (m<<10)|0x1f)
__device__ __forceinline__ float xmax16(float v) {
    v = fmaxf(v, __int_as_float(__builtin_amdgcn_ds_swizzle(__float_as_int(v), 0x041f)));
    v = fmaxf(v, __int_as_float(__builtin_amdgcn_ds_swizzle(__float_as_int(v), 0x081f)));
    v = fmaxf(v, __int_as_float(__builtin_amdgcn_ds_swizzle(__float_as_int(v), 0x101f)));
    v = fmaxf(v, __int_as_float(__builtin_amdgcn_ds_swizzle(__float_as_int(v), 0x201f)));
    return v;
}

// ---------------------------------------------------------------- conv ----
// One block = one (window, 64-wide time tile).  Both convs run on
// v_wmma_f32_16x16x32_f16; wave w owns M-tile (channel rows) w*16..w*16+15.
__global__ void __launch_bounds__(256)
conv_kernel(const float* __restrict__ batch, const float* __restrict__ train,
            const float* __restrict__ b2,
            const _Float16* __restrict__ Afrag2, const _Float16* __restrict__ Afrag1,
            const int* __restrict__ wdesc, float* __restrict__ Hmax) {
    const int w    = blockIdx.x;
    const int tile = blockIdx.y;
    const int kind = wdesc[w * 4 + 0];
    const int row  = wdesc[w * 4 + 1];
    const int st   = wdesc[w * 4 + 2];
    const int len  = wdesc[w * 4 + 3];
    if (tile * TILE >= len) return;
    const int t0  = tile * TILE;
    const int tid = threadIdx.x;

    __shared__ float    xs[CC * 72];              // input tile, halo 4 each side
    __shared__ _Float16 xim[80 * XIROW];          // conv1 im2col panel (f16)
    __shared__ _Float16 h1s[(TILE + 4) * H1ROW];  // conv1 output panel (f16)
    __shared__ float    bs2[HH];

    const float* src = (kind ? train : batch) + (size_t)row * (CC * LL);

    // stage x tile (zero outside window -> SAME padding of conv1) + conv2 bias
    for (int i = tid; i < CC * 72; i += 256) {
        int c = i / 72, j = i % 72;
        int wt = t0 - 4 + j;
        float v = 0.0f;
        if (wt >= 0 && wt < len) v = src[c * LL + st + wt];
        xs[i] = v;
    }
    if (tid < HH) bs2[tid] = b2[tid];
    __syncthreads();

    // f16 im2col panel: xim[t][c*5+k] = x[c][t+k-2]; col 50 = 1.0 (bias input)
    for (int i = tid; i < 80 * 64; i += 256) {
        int t = i >> 6, kidx = i & 63;
        float v = 0.0f;
        if (kidx == CK1)                        v = 1.0f;
        else if (t < TILE + 4 && kidx < CK1) {
            int c = kidx / KK, k = kidx % KK;
            v = xs[c * 72 + t + k];
        }
        xim[t * XIROW + kidx] = (_Float16)v;
    }
    __syncthreads();

    const int lane   = tid & 31;
    const int mt     = tid >> 5;                  // wave id = M tile
    const int hihalf = (lane < 16) ? 0 : 8;
    const int n      = lane & 15;
    const int hbase  = mt * 16 + ((lane < 16) ? 0 : 8);

    // ---- conv1 via WMMA: [128 x 64(pad+bias)] x [64 x 80] -> h1s[t][h] ----
    {
        v16h a0 = *(const v16h*)(Afrag1 + ((size_t)(mt * 2 + 0) * 32 + lane) * 16);
        v16h a1 = *(const v16h*)(Afrag1 + ((size_t)(mt * 2 + 1) * 32 + lane) * 16);
        #pragma unroll
        for (int nt = 0; nt < 5; ++nt) {
            const _Float16* bp = &xim[(nt * 16 + n) * XIROW + hihalf];
            union { v16h v; v8h h2[2]; } u0, u1;
            u0.h2[0] = *(const v8h*)(bp);       u0.h2[1] = *(const v8h*)(bp + 16);
            u1.h2[0] = *(const v8h*)(bp + 32);  u1.h2[1] = *(const v8h*)(bp + 48);
            v8f c1 = {};
            c1 = __builtin_amdgcn_wmma_f32_16x16x32_f16(false, a0, false, u0.v,
                                                        (short)0, c1, false, false);
            c1 = __builtin_amdgcn_wmma_f32_16x16x32_f16(false, a1, false, u1.v,
                                                        (short)0, c1, false, false);
            const int t  = nt * 16 + n;           // local h1 row (window time t0-2+t)
            const int wt = t0 - 2 + t;
            const bool inwin = (wt >= 0) && (wt < len);  // conv2 pads conv1 OUTPUT
            union { v8h v; int i4[4]; _Float16 e[8]; } o;
            #pragma unroll
            for (int r = 0; r < 8; ++r)
                o.e[r] = (_Float16)fmaxf(c1[r], 0.0f);   // bias already in GEMM
            #pragma unroll
            for (int j = 0; j < 4; ++j)                  // branch-free zero select
                o.i4[j] = inwin ? o.i4[j] : 0;
            if (t < TILE + 4)
                *(v8h*)&h1s[t * H1ROW + hbase] = o.v;    // 8 consecutive channels
        }
    }
    __syncthreads();

    // ---- conv2 via WMMA: [128 x 640] x [640 x 64], K reordered k*128+c ----
    v8f acc[4] = {};
    for (int kb = 0; kb < NKB; ++kb) {
        v16h a = *(const v16h*)(Afrag2 + ((size_t)(mt * NKB + kb) * 32 + lane) * 16);
        const int k  = kb >> 2;                   // conv tap (constant per kb)
        const int cb = (kb & 3) * 32 + hihalf;    // channel base for this lane half
        #pragma unroll
        for (int nt = 0; nt < 4; ++nt) {
            const _Float16* bp = &h1s[(nt * 16 + n + k) * H1ROW + cb];
            union { v16h v; v8h h2[2]; } u;
            u.h2[0] = *(const v8h*)(bp);
            u.h2[1] = *(const v8h*)(bp + 16);
            acc[nt] = __builtin_amdgcn_wmma_f32_16x16x32_f16(
                false, a, false, u.v, (short)0, acc[nt], false, false);
        }
    }

    // epilogue: bias + ReLU + max over time, pooled via atomicMax (vals >= 0)
    float bias2[8];
    #pragma unroll
    for (int r = 0; r < 8; ++r) bias2[r] = bs2[hbase + r];
    #pragma unroll
    for (int r = 0; r < 8; ++r) {
        float m = 0.0f;
        #pragma unroll
        for (int nt = 0; nt < 4; ++nt)
            m = fmaxf(m, fmaxf(acc[nt][r] + bias2[r], 0.0f));
        m = xmax16(m);
        if ((lane & 15) == 0)
            atomicMax((int*)&Hmax[(size_t)w * HH + hbase + r], __float_as_int(m));
    }
}

// ------------------------------------------------------------------ FC ----
__global__ void fc_kernel(const float* __restrict__ Hmax, const float* __restrict__ Wo,
                          const float* __restrict__ bo, float* __restrict__ Z) {
    int w = blockIdx.x;
    int d = threadIdx.x;                      // 160
    __shared__ float hrow[HH];
    for (int i = threadIdx.x; i < HH; i += DD) hrow[i] = Hmax[(size_t)w * HH + i];
    __syncthreads();
    float acc = bo[d];
    #pragma unroll 4
    for (int h = 0; h < HH; ++h) acc = fmaf(hrow[h], Wo[h * DD + d], acc);
    Z[(size_t)w * DD + d] = acc;
}

// ---------------------------------------------------------------- loss ----
__global__ void loss_kernel(const float* __restrict__ Z, float* __restrict__ out) {
    int b = threadIdx.x;                      // 128
    const float* zb = Z + (size_t)b * DD;
    const float* zp = Z + (size_t)(BB + b) * DD;
    float pos = 0.0f, ng[NEG] = {0.0f, 0.0f, 0.0f, 0.0f};
    for (int i = 0; i < DD; ++i) {
        float z = zb[i];
        pos = fmaf(z, zp[i], pos);
        #pragma unroll
        for (int nn = 0; nn < NEG; ++nn)
            ng[nn] = fmaf(z, Z[(size_t)(2 * BB + nn * BB + b) * DD + i], ng[nn]);
    }
    float m = pos;
    #pragma unroll
    for (int nn = 0; nn < NEG; ++nn) m = fmaxf(m, ng[nn]);
    float s = expf(pos - m);
    #pragma unroll
    for (int nn = 0; nn < NEG; ++nn) s += expf(ng[nn] - m);
    float loss = m + logf(s) - pos;           // = -log_softmax[:,0]

    __shared__ float red[BB];
    red[b] = loss;
    __syncthreads();
    for (int sft = BB / 2; sft > 0; sft >>= 1) {
        if (b < sft) red[b] += red[b + sft];
        __syncthreads();
    }
    if (b == 0) out[0] = red[0] * (1.0f / (float)BB);
}

// -------------------------------------------------------------- launch ----
extern "C" void kernel_launch(void* const* d_in, const int* in_sizes, int n_in,
                              void* d_out, int out_size, void* d_ws, size_t ws_size,
                              hipStream_t stream) {
    (void)in_sizes; (void)n_in; (void)out_size; (void)ws_size;
    const float* batch     = (const float*)d_in[0];
    const float* train     = (const float*)d_in[1];
    const float* W1        = (const float*)d_in[2];
    const float* b1        = (const float*)d_in[3];
    const float* W2        = (const float*)d_in[4];
    const float* b2        = (const float*)d_in[5];
    const float* Wo        = (const float*)d_in[6];
    const float* bo        = (const float*)d_in[7];
    const int*   samples   = (const int*)d_in[8];
    const int*   start_sel = (const int*)d_in[9];
    const int*   start_pos = (const int*)d_in[10];
    const int*   start_neg = (const int*)d_in[11];
    float*       out       = (float*)d_out;

    char* ws = (char*)d_ws;
    size_t off = 0;
    float*    Hmax   = (float*)(ws + off);    off += (size_t)NWIN * HH * sizeof(float);   // 384 KB
    _Float16* Afrag2 = (_Float16*)(ws + off); off += (size_t)8 * NKB * 32 * 16 * 2;       // 160 KB
    _Float16* Afrag1 = (_Float16*)(ws + off); off += (size_t)8 * 2 * 32 * 16 * 2;         //  16 KB
    int*      wdesc  = (int*)(ws + off);      off += (size_t)NWIN * 4 * sizeof(int);      //  12 KB
    float*    Z      = (float*)(ws + off);    off += (size_t)NWIN * DD * sizeof(float);   // 480 KB

    setup_kernel<<<(NWIN * HH + 255) / 256, 256, 0, stream>>>(
        samples, start_sel, start_pos, start_neg, wdesc, Hmax);
    afrag_kernel<<<(8 * NKB * 32 + 8 * 2 * 32 + 255) / 256, 256, 0, stream>>>(
        W2, W1, b1, Afrag2, Afrag1);
    conv_kernel<<<dim3(NWIN, LSEL / TILE), 256, 0, stream>>>(
        batch, train, b2, Afrag2, Afrag1, wdesc, Hmax);
    fc_kernel<<<NWIN, DD, 0, stream>>>(Hmax, Wo, bo, Z);
    loss_kernel<<<1, BB, 0, stream>>>(Z, out);
}
